// parts_query_encoder_decoder_37958920962511
// MI455X (gfx1250) — matrix operand order repaired and checked by reading, not compiled, not run on hardware
//
#include <hip/hip_runtime.h>
#include <math.h>

// ---------------------------------------------------------------------------
// CDNA5 (gfx1250) bf16 WMMA transformer forward.
// wave32; WMMA 16x16x32 bf16 -> f32 accumulate; async global->LDS staging.
// ---------------------------------------------------------------------------

typedef __attribute__((ext_vector_type(16))) __bf16 v16bf;
typedef __attribute__((ext_vector_type(8)))  __bf16 v8bf;
typedef __attribute__((ext_vector_type(4)))  __bf16 v4bf;
typedef __attribute__((ext_vector_type(8)))  float  v8f;

#define FUSE_BIAS  1
#define FUSE_RESID 2
#define FUSE_GELU  4

__device__ __forceinline__ v8f wmma_bf16(v16bf a, v16bf b, v8f c) {
  return __builtin_amdgcn_wmma_f32_16x16x32_bf16(
      /*neg_a=*/false, a, /*neg_b=*/false, b,
      /*c_mod=*/(short)0, c, /*reuse_a=*/false, /*reuse_b=*/false);
}

// Async 16B copy global -> LDS (GLOBAL_LOAD_ASYNC_TO_LDS_B128, ASYNCcnt).
__device__ __forceinline__ void async_ld_b128(const void* gaddr, void* lds) {
  const unsigned off = (unsigned)(uintptr_t)lds;  // low 32b of LDS generic ptr
  asm volatile("global_load_async_to_lds_b128 %0, %1, off"
               :: "v"(off), "v"(gaddr)
               : "memory");
}
__device__ __forceinline__ void wait_async0() {
  asm volatile("s_wait_asynccnt 0x0" ::: "memory");
}

// A fragment (16x32, bf16): lane half h, m=lane&15 -> row m,
// K = h*8..h*8+7 and 16+h*8..16+h*8+7  (ISA 7.12.2 "16-bit A-Matrix 16x32")
__device__ __forceinline__ v16bf frag_a(const __bf16* base, int stride) {
  const int lane = threadIdx.x & 31;
  const int h = lane >> 4, m = lane & 15;
  const __bf16* p = base + m * stride + h * 8;
  v8bf lo = *(const v8bf*)(p);
  v8bf hi = *(const v8bf*)(p + 16);
  v16bf f;
#pragma unroll
  for (int i = 0; i < 8; ++i) { f[i] = lo[i]; f[i + 8] = hi[i]; }
  return f;
}

// B fragment (32x16, bf16) read from a TRANSPOSED tile [N][K]:
// lane half h, n=lane&15 -> column n, K = h*8..h*8+7 and 16+h*8..16+h*8+7.
__device__ __forceinline__ v16bf frag_b(const __bf16* base, int stride) {
  const int lane = threadIdx.x & 31;
  const int h = lane >> 4, n = lane & 15;
  const __bf16* p = base + n * stride + h * 8;
  v8bf lo = *(const v8bf*)(p);
  v8bf hi = *(const v8bf*)(p + 16);
  v16bf f;
#pragma unroll
  for (int i = 0; i < 8; ++i) { f[i] = lo[i]; f[i + 8] = hi[i]; }
  return f;
}

__device__ __forceinline__ float gelu_exact(float v) {
  return 0.5f * v * (1.0f + erff(v * 0.70710678118654752f));
}

// ---------------------------------------------------------------------------
// GEMM: C[M,N] = A[M,K](bf16) @ B[K,N](bf16) (+bias)(+gelu)(+resid), f32 out.
// Block: 128 threads (4 waves), 64x64 output tile, K-steps of 64
// (8 WMMA per staging round). A tile staged via async global->LDS.
// ---------------------------------------------------------------------------
__global__ void __launch_bounds__(128)
gemm_bf16_kernel(const __bf16* __restrict__ A, const __bf16* __restrict__ B,
                 const float* __restrict__ bias, const float* __restrict__ resid,
                 float* __restrict__ C, int M, int N, int K, int fuse) {
  __shared__ __align__(16) __bf16 As[64][72];  // [M][K] tile, pad to 72
  __shared__ __align__(16) __bf16 Bt[64][72];  // transposed [N][K] tile

  const int tid = threadIdx.x;
  const int wave = tid >> 5;
  const int rW = (wave >> 1) * 32;
  const int cW = (wave & 1) * 32;
  const int m0 = blockIdx.y * 64;
  const int n0 = blockIdx.x * 64;

  const v8f vzero = {0.f, 0.f, 0.f, 0.f, 0.f, 0.f, 0.f, 0.f};
  v8f acc[2][2];
#pragma unroll
  for (int i = 0; i < 2; ++i)
#pragma unroll
    for (int j = 0; j < 2; ++j) acc[i][j] = vzero;

  for (int k0 = 0; k0 < K; k0 += 64) {
    // Stage A tile 64x64 asynchronously (global -> LDS, no VGPR round-trip).
#pragma unroll
    for (int i = 0; i < 4; ++i) {
      const int v = tid * 4 + i;           // 0..511
      const int row = v >> 3;              // 0..63
      const int kc = (v & 7) * 8;          // 0..56
      async_ld_b128(A + (size_t)(m0 + row) * K + (k0 + kc), &As[row][kc]);
    }
    // Stage B tile transposed: load 8 along N, scatter into Bt[n][k].
#pragma unroll
    for (int i = 0; i < 4; ++i) {
      const int v = tid * 4 + i;
      const int kr = v >> 3;               // 0..63
      const int nc = (v & 7) * 8;          // 0..56
      v8bf t = *(const v8bf*)(B + (size_t)(k0 + kr) * N + (n0 + nc));
#pragma unroll
      for (int e = 0; e < 8; ++e) Bt[nc + e][kr] = t[e];
    }
    if (k0 + 64 < K)
      __builtin_prefetch(A + (size_t)(m0 + (tid >> 1)) * K + (k0 + 64), 0, 3);
    wait_async0();
    __syncthreads();

#pragma unroll
    for (int kk = 0; kk < 64; kk += 32) {
      v16bf a0 = frag_a(&As[rW][kk], 72);
      v16bf a1 = frag_a(&As[rW + 16][kk], 72);
      v16bf b0 = frag_b(&Bt[cW][kk], 72);
      v16bf b1 = frag_b(&Bt[cW + 16][kk], 72);
      acc[0][0] = wmma_bf16(a0, b0, acc[0][0]);
      acc[0][1] = wmma_bf16(a0, b1, acc[0][1]);
      acc[1][0] = wmma_bf16(a1, b0, acc[1][0]);
      acc[1][1] = wmma_bf16(a1, b1, acc[1][1]);
    }
    __syncthreads();
  }

  const int lane = tid & 31;
  const int h = lane >> 4, ml = lane & 15;
#pragma unroll
  for (int i = 0; i < 2; ++i)
#pragma unroll
    for (int j = 0; j < 2; ++j)
#pragma unroll
      for (int r = 0; r < 8; ++r) {
        const int row = m0 + rW + i * 16 + h * 8 + r;
        const int col = n0 + cW + j * 16 + ml;
        float v = acc[i][j][r];
        if (fuse & FUSE_BIAS) v += bias[col];
        if (fuse & FUSE_GELU) v = gelu_exact(v);
        if (fuse & FUSE_RESID) v += resid[(size_t)row * N + col];
        C[(size_t)row * N + col] = v;
      }
}

// ---------------------------------------------------------------------------
// Flash attention: Q[B,Lq,256], K/V[B,Lk,256] bf16 (head h at cols h*64..),
// O[B,Lq,256] f32.  Block = 64 queries, iterate keys in tiles of 64.
// grid = (Lq/64, H, B); 128 threads.  Q/K tiles staged via async copies.
// ---------------------------------------------------------------------------
__global__ void __launch_bounds__(128)
flash_attn_kernel(const __bf16* __restrict__ Q, const __bf16* __restrict__ K,
                  const __bf16* __restrict__ V, float* __restrict__ O,
                  int Lq, int Lk, float scale) {
  const int D = 256;
  __shared__ __align__(16) __bf16 Qs[64][72];
  __shared__ __align__(16) __bf16 Ks[64][72];
  __shared__ __align__(16) __bf16 Vt[64][72];   // [d][key]
  __shared__ __align__(16) __bf16 Ps[64][72];
  __shared__ __align__(16) float  Sf[64][72];
  __shared__ float rm[64], rl[64], rf[64];

  const int tid = threadIdx.x;
  const int wave = tid >> 5;
  const int lane = tid & 31;
  const int h16 = lane >> 4, ml = lane & 15;
  const int rW = (wave >> 1) * 32;
  const int cW = (wave & 1) * 32;
  const int qb = blockIdx.x, head = blockIdx.y, batch = blockIdx.z;

  const size_t qbase = (size_t)batch * Lq * D + (size_t)head * 64;
  const size_t kbase = (size_t)batch * Lk * D + (size_t)head * 64;

  // Async-stage the Q tile; completion covered by the wait inside iter 0.
#pragma unroll
  for (int i = 0; i < 4; ++i) {
    const int v = tid * 4 + i;     // 0..511
    const int row = v >> 3;        // 0..63
    const int dc = (v & 7) * 8;    // 0..56
    async_ld_b128(Q + qbase + (size_t)(qb * 64 + row) * D + dc, &Qs[row][dc]);
  }
  if (tid < 64) { rm[tid] = -1e30f; rl[tid] = 0.f; }

  const v8f vzero = {0.f, 0.f, 0.f, 0.f, 0.f, 0.f, 0.f, 0.f};
  v8f oacc[2][2];
#pragma unroll
  for (int i = 0; i < 2; ++i)
#pragma unroll
    for (int j = 0; j < 2; ++j) oacc[i][j] = vzero;

  for (int kb = 0; kb < Lk; kb += 64) {
    __syncthreads();  // previous-iter LDS consumers done
#pragma unroll
    for (int i = 0; i < 4; ++i) {
      const int v = tid * 4 + i;
      const int row = v >> 3;
      const int dc = (v & 7) * 8;
      async_ld_b128(K + kbase + (size_t)(kb + row) * D + dc, &Ks[row][dc]);
      v8bf t = *(const v8bf*)(V + kbase + (size_t)(kb + row) * D + dc);
#pragma unroll
      for (int e = 0; e < 8; ++e) Vt[dc + e][row] = t[e];
    }
    wait_async0();
    __syncthreads();

    // S = Q @ K^T  (dh = 64 -> two K-steps of 32)
    v8f s[2][2];
#pragma unroll
    for (int i = 0; i < 2; ++i)
#pragma unroll
      for (int j = 0; j < 2; ++j) s[i][j] = vzero;
#pragma unroll
    for (int kk = 0; kk < 64; kk += 32) {
      v16bf a0 = frag_a(&Qs[rW][kk], 72);
      v16bf a1 = frag_a(&Qs[rW + 16][kk], 72);
      v16bf b0 = frag_b(&Ks[cW][kk], 72);
      v16bf b1 = frag_b(&Ks[cW + 16][kk], 72);
      s[0][0] = wmma_bf16(a0, b0, s[0][0]);
      s[0][1] = wmma_bf16(a0, b1, s[0][1]);
      s[1][0] = wmma_bf16(a1, b0, s[1][0]);
      s[1][1] = wmma_bf16(a1, b1, s[1][1]);
    }
#pragma unroll
    for (int i = 0; i < 2; ++i)
#pragma unroll
      for (int j = 0; j < 2; ++j)
#pragma unroll
        for (int r = 0; r < 8; ++r)
          Sf[rW + i * 16 + h16 * 8 + r][cW + j * 16 + ml] = s[i][j][r] * scale;
    __syncthreads();

    // Online softmax (one thread per query row).
    if (tid < 64) {
      float mx = rm[tid];
      for (int j = 0; j < 64; ++j) mx = fmaxf(mx, Sf[tid][j]);
      const float f = __expf(rm[tid] - mx);
      float sum = 0.f;
      for (int j = 0; j < 64; ++j) {
        const float p = __expf(Sf[tid][j] - mx);
        Ps[tid][j] = (__bf16)p;
        sum += p;
      }
      rl[tid] = rl[tid] * f + sum;
      rm[tid] = mx;
      rf[tid] = f;
    }
    __syncthreads();

    // O = O*f + P @ V
#pragma unroll
    for (int i = 0; i < 2; ++i)
#pragma unroll
      for (int j = 0; j < 2; ++j)
#pragma unroll
        for (int r = 0; r < 8; ++r)
          oacc[i][j][r] *= rf[rW + i * 16 + h16 * 8 + r];
#pragma unroll
    for (int kk = 0; kk < 64; kk += 32) {
      v16bf a0 = frag_a(&Ps[rW][kk], 72);
      v16bf a1 = frag_a(&Ps[rW + 16][kk], 72);
      v16bf b0 = frag_b(&Vt[cW][kk], 72);
      v16bf b1 = frag_b(&Vt[cW + 16][kk], 72);
      oacc[0][0] = wmma_bf16(a0, b0, oacc[0][0]);
      oacc[0][1] = wmma_bf16(a0, b1, oacc[0][1]);
      oacc[1][0] = wmma_bf16(a1, b0, oacc[1][0]);
      oacc[1][1] = wmma_bf16(a1, b1, oacc[1][1]);
    }
  }
  __syncthreads();

#pragma unroll
  for (int i = 0; i < 2; ++i)
#pragma unroll
    for (int j = 0; j < 2; ++j)
#pragma unroll
      for (int r = 0; r < 8; ++r) {
        const int row = rW + i * 16 + h16 * 8 + r;
        const int col = cW + j * 16 + ml;
        O[qbase + (size_t)(qb * 64 + row) * D + col] = oacc[i][j][r] / rl[row];
      }
}

// ---------------------------------------------------------------------------
// Fused (pos-add ->) LayerNorm(D=256) -> bf16.  One wave per row.
// pos (may be null) indexed [token*64 + (d&63)] (qpos/kpos layout).
// ---------------------------------------------------------------------------
__global__ void __launch_bounds__(256)
ln_bf16_kernel(const float* __restrict__ X, const float* __restrict__ g,
               const float* __restrict__ bta, const float* __restrict__ pos,
               __bf16* __restrict__ Y, int rows, int tokens) {
  const int D = 256;
  const int row = blockIdx.x * 8 + (threadIdx.x >> 5);
  const int lane = threadIdx.x & 31;
  if (row >= rows) return;
  const int n = row % tokens;

  float vals[8];
  float s = 0.f, sq = 0.f;
#pragma unroll
  for (int i = 0; i < 8; ++i) {
    const int d = lane + i * 32;
    float v = X[(size_t)row * D + d];
    if (pos) v += pos[n * 64 + (d & 63)];
    vals[i] = v;
    s += v;
    sq += v * v;
  }
#pragma unroll
  for (int off = 16; off; off >>= 1) {
    s += __shfl_xor(s, off);
    sq += __shfl_xor(sq, off);
  }
  const float mean = s * (1.f / 256.f);
  const float var = sq * (1.f / 256.f) - mean * mean;
  const float rstd = rsqrtf(var + 1e-5f);
#pragma unroll
  for (int i = 0; i < 8; ++i) {
    const int d = lane + i * 32;
    Y[(size_t)row * D + d] = (__bf16)((vals[i] - mean) * rstd * g[d] + bta[d]);
  }
}

__global__ void __launch_bounds__(256)
cvt_f32_bf16_kernel(const float* __restrict__ X, __bf16* __restrict__ Y, long n) {
  const long i = ((long)blockIdx.x * 256 + threadIdx.x) * 4;
  if (i + 3 < n) {
    const float4 v = *(const float4*)(X + i);
    v4bf o;
    o[0] = (__bf16)v.x; o[1] = (__bf16)v.y; o[2] = (__bf16)v.z; o[3] = (__bf16)v.w;
    *(v4bf*)(Y + i) = o;
  }
}

__global__ void __launch_bounds__(256)
bcast_parts_kernel(const float* __restrict__ src, float* __restrict__ dst,
                   int per, int total) {
  const int i = blockIdx.x * 256 + threadIdx.x;
  if (i < total) dst[i] = src[i % per];
}

// ---------------------------------------------------------------------------
// Host orchestration
// ---------------------------------------------------------------------------
extern "C" void kernel_launch(void* const* d_in, const int* in_sizes, int n_in,
                              void* d_out, int out_size, void* d_ws, size_t ws_size,
                              hipStream_t stream) {
  (void)in_sizes; (void)n_in; (void)out_size; (void)ws_size;
  const int B_ = 4, L_ = 3136, D_ = 256, H_ = 4, NP_ = 64;
  const int Mx = B_ * L_;   // 12544
  const int Mp = B_ * NP_;  // 256

  auto in = [&](int i) { return (const float*)d_in[i]; };
  // setup_inputs() insertion-order flattening:
  // 0:x 1:parts 2:qpos 3:kpos
  // enc.attn: 4..14 (nq_g,nq_b,nk_g,nk_b,nv_g,nv_b,wq,wk,wv,wp,bp)
  // 15:r_ln_g 16:r_ln_b 17:r_w
  // enc.ffn: 18..23 (ln_g,ln_b,w1,b1,w2,b2)
  // dec.attn1: 24..34 ; dec.ffn1: 35..40 ; dec.mhsa: 41..48 ; dec.ffn2: 49..54
  const float* x = in(0);

  char* wsp = (char*)d_ws;
  auto carve = [&](size_t bytes) -> void* {
    void* r = (void*)wsp;
    wsp += (bytes + 255) & ~(size_t)255;
    return r;
  };
  auto launch_cvt = [&](const float* src, __bf16* dst, long n) {
    cvt_f32_bf16_kernel<<<dim3((unsigned)((n / 4 + 255) / 256)), dim3(256), 0,
                          stream>>>(src, dst, n);
  };
  auto conv_w = [&](int idx, long n) -> __bf16* {
    __bf16* dst = (__bf16*)carve((size_t)n * sizeof(__bf16));
    launch_cvt(in(idx), dst, n);
    return dst;
  };
  auto gemm = [&](const __bf16* A, const __bf16* Bm, const float* bias,
                  const float* resid, float* C, int M, int N, int K, int fuse) {
    gemm_bf16_kernel<<<dim3(N / 64, M / 64), dim3(128), 0, stream>>>(
        A, Bm, bias, resid, C, M, N, K, fuse);
  };
  auto ln = [&](const float* X, const float* g, const float* b,
                const float* pos, __bf16* Y, int rows, int tokens) {
    ln_bf16_kernel<<<dim3(rows / 8), dim3(256), 0, stream>>>(X, g, b, pos, Y,
                                                             rows, tokens);
  };
  auto flash = [&](const __bf16* Qp, const __bf16* Kp, const __bf16* Vp,
                   float* Op, int Lq, int Lk) {
    flash_attn_kernel<<<dim3(Lq / 64, H_, B_), dim3(128), 0, stream>>>(
        Qp, Kp, Vp, Op, Lq, Lk, 0.125f);
  };

  const long DD = (long)D_ * D_;          // 65536
  const long DH3 = (long)D_ * 3 * D_;     // 196608

  // bf16 weights
  __bf16 *e_wq = conv_w(10, DD), *e_wk = conv_w(11, DD), *e_wv = conv_w(12, DD),
         *e_wp = conv_w(13, DD);
  __bf16* r_w = conv_w(17, DD);
  __bf16 *e_w1 = conv_w(20, DD), *e_w2 = conv_w(22, DD);
  __bf16 *a_wq = conv_w(30, DD), *a_wk = conv_w(31, DD), *a_wv = conv_w(32, DD),
         *a_wp = conv_w(33, DD);
  __bf16 *f1_w1 = conv_w(37, DH3), *f1_w2 = conv_w(39, DH3);
  __bf16 *m_wq = conv_w(41, DD), *m_wk = conv_w(43, DD), *m_wv = conv_w(45, DD),
         *m_wo = conv_w(47, DD);
  __bf16 *f2_w1 = conv_w(51, DH3), *f2_w2 = conv_w(53, DH3);

  // activations
  float* parts_cur = (float*)carve((size_t)Mp * D_ * 4);
  float* parts2    = (float*)carve((size_t)Mp * D_ * 4);
  float* parts3    = (float*)carve((size_t)Mp * D_ * 4);
  float* parts4    = (float*)carve((size_t)Mp * D_ * 4);
  __bf16* lnbuf    = (__bf16*)carve((size_t)Mx * D_ * 2);
  __bf16* qbf      = (__bf16*)carve((size_t)Mx * D_ * 2);
  __bf16* kbf      = (__bf16*)carve((size_t)Mx * D_ * 2);
  __bf16* vbf      = (__bf16*)carve((size_t)Mx * D_ * 2);
  float* gtmp      = (float*)carve((size_t)Mx * 768 * 4);
  __bf16* hbf      = (__bf16*)carve((size_t)Mx * 768 * 2);
  float* attno     = (float*)carve((size_t)Mx * D_ * 4);
  __bf16* attno_bf = (__bf16*)carve((size_t)Mx * D_ * 2);
  float* out1      = (float*)carve((size_t)Mx * D_ * 4);
  float* out2      = (float*)carve((size_t)Mx * D_ * 4);
  float* out3      = (float*)carve((size_t)Mx * D_ * 4);

  // parts = broadcast(params.parts)
  bcast_parts_kernel<<<dim3((Mp * D_ + 255) / 256), dim3(256), 0, stream>>>(
      in(1), parts_cur, NP_ * D_, Mp * D_);

  // ===== Encoder cross-attention: parts <- attn(parts+qpos, x, x) =====
  ln(parts_cur, in(4), in(5), in(2) /*qpos*/, lnbuf, Mp, NP_);
  gemm(lnbuf, e_wq, nullptr, nullptr, gtmp, Mp, D_, D_, 0);
  launch_cvt(gtmp, qbf, (long)Mp * D_);
  ln(x, in(6), in(7), nullptr, lnbuf, Mx, L_);
  gemm(lnbuf, e_wk, nullptr, nullptr, gtmp, Mx, D_, D_, 0);
  launch_cvt(gtmp, kbf, (long)Mx * D_);
  ln(x, in(8), in(9), nullptr, lnbuf, Mx, L_);
  gemm(lnbuf, e_wv, nullptr, nullptr, gtmp, Mx, D_, D_, 0);
  launch_cvt(gtmp, vbf, (long)Mx * D_);
  flash(qbf, kbf, vbf, attno, NP_, L_);
  launch_cvt(attno, attno_bf, (long)Mp * D_);
  gemm(attno_bf, e_wp, in(14), parts_cur, parts2, Mp, D_, D_,
       FUSE_BIAS | FUSE_RESID);

  // SimpleReasoning: parts += LN(parts) @ r_w
  ln(parts2, in(15), in(16), nullptr, lnbuf, Mp, NP_);
  gemm(lnbuf, r_w, nullptr, parts2, parts3, Mp, D_, D_, FUSE_RESID);

  // Encoder FFN (hidden = 256)
  ln(parts3, in(18), in(19), nullptr, lnbuf, Mp, NP_);
  gemm(lnbuf, e_w1, in(21), nullptr, gtmp, Mp, D_, D_, FUSE_BIAS | FUSE_GELU);
  launch_cvt(gtmp, hbf, (long)Mp * D_);
  gemm(hbf, e_w2, in(23), parts3, parts4, Mp, D_, D_, FUSE_BIAS | FUSE_RESID);

  // ===== Decoder attn1: out1 = x + attn(x, parts+kpos, parts) =====
  ln(x, in(24), in(25), nullptr, lnbuf, Mx, L_);
  gemm(lnbuf, a_wq, nullptr, nullptr, gtmp, Mx, D_, D_, 0);
  launch_cvt(gtmp, qbf, (long)Mx * D_);
  ln(parts4, in(26), in(27), in(3) /*kpos*/, lnbuf, Mp, NP_);
  gemm(lnbuf, a_wk, nullptr, nullptr, gtmp, Mp, D_, D_, 0);
  launch_cvt(gtmp, kbf, (long)Mp * D_);
  ln(parts4, in(28), in(29), nullptr, lnbuf, Mp, NP_);
  gemm(lnbuf, a_wv, nullptr, nullptr, gtmp, Mp, D_, D_, 0);
  launch_cvt(gtmp, vbf, (long)Mp * D_);
  flash(qbf, kbf, vbf, attno, L_, NP_);
  launch_cvt(attno, attno_bf, (long)Mx * D_);
  gemm(attno_bf, a_wp, in(34), x, out1, Mx, D_, D_, FUSE_BIAS | FUSE_RESID);

  // FFN1 (hidden = 768)
  ln(out1, in(35), in(36), nullptr, lnbuf, Mx, L_);
  gemm(lnbuf, f1_w1, in(38), nullptr, gtmp, Mx, 768, D_, FUSE_BIAS | FUSE_GELU);
  launch_cvt(gtmp, hbf, (long)Mx * 768);
  gemm(hbf, f1_w2, in(40), out1, out2, Mx, D_, 768, FUSE_BIAS | FUSE_RESID);

  // ===== MHSA on out2, residual from x =====
  launch_cvt(out2, lnbuf, (long)Mx * D_);
  gemm(lnbuf, m_wq, in(42), nullptr, gtmp, Mx, D_, D_, FUSE_BIAS);
  launch_cvt(gtmp, qbf, (long)Mx * D_);
  gemm(lnbuf, m_wk, in(44), nullptr, gtmp, Mx, D_, D_, FUSE_BIAS);
  launch_cvt(gtmp, kbf, (long)Mx * D_);
  gemm(lnbuf, m_wv, in(46), nullptr, gtmp, Mx, D_, D_, FUSE_BIAS);
  launch_cvt(gtmp, vbf, (long)Mx * D_);
  flash(qbf, kbf, vbf, attno, L_, L_);
  launch_cvt(attno, attno_bf, (long)Mx * D_);
  gemm(attno_bf, m_wo, in(48), x, out3, Mx, D_, D_, FUSE_BIAS | FUSE_RESID);

  // FFN2 -> d_out
  ln(out3, in(49), in(50), nullptr, lnbuf, Mx, L_);
  gemm(lnbuf, f2_w1, in(52), nullptr, gtmp, Mx, 768, D_, FUSE_BIAS | FUSE_GELU);
  launch_cvt(gtmp, hbf, (long)Mx * 768);
  gemm(hbf, f2_w2, in(54), out3, (float*)d_out, Mx, D_, 768,
       FUSE_BIAS | FUSE_RESID);
}
